// SCLSTM_76218489634840
// MI455X (gfx1250) — compile-verified, matrix operand order checked by reading, not run on hardware
//
#include <hip/hip_runtime.h>
#include <hip/hip_bf16.h>

// ---------------- problem constants ----------------
#define T_STEPS 256
#define BATCH   64
#define IN_DIM  512
#define H_DIM   1024
#define DA_DIM  128
#define V_DIM   2048
#define KCAT    1536            // IN + H
#define NCAT    4224            // 4H (gates) + DA (r-preact)
#define ETA_C   1e-4f
#define KEXI_C  1.5f

// ---------------- WMMA / descriptor types ----------------
typedef __attribute__((ext_vector_type(16))) __bf16        v16bf;
typedef __attribute__((ext_vector_type(8)))  float         v8f;
typedef __attribute__((ext_vector_type(4)))  unsigned int  u32x4;
typedef __attribute__((ext_vector_type(8)))  int           i32x8;
typedef __attribute__((ext_vector_type(4)))  int           i32x4;

union FragBF { v16bf v; u32x4 q[2]; };

__device__ __forceinline__ v16bf load_frag(const __bf16* p) {
  // 16 bf16: K 0..7 (or 8..15 in hi lanes) then K 16..23 (or 24..31)
  FragBF f;
  f.q[0] = *(const u32x4*)(p);
  f.q[1] = *(const u32x4*)(p + 16);
  return f.v;
}

__device__ __forceinline__ v8f vzero8() {
  v8f z = {0.f, 0.f, 0.f, 0.f, 0.f, 0.f, 0.f, 0.f};
  return z;
}

__device__ __forceinline__ float sigmoidf_(float x) {
  return 1.0f / (1.0f + __expf(-x));
}

// Issue one TDM 2-D tile load: [rows x cols] bf16, global(row stride) -> LDS packed.
// D# per CDNA5 ISA ch.8: group0 = {count, lds_addr, global_addr[56:0], type=2},
// group1 = {data_size=2B, tensor_dim0/1, tile_dim0/1, tensor_dim0_stride}.
// This toolchain's builtin takes 6 args (g0, g1, g2, g3, g_extra, cpol).
__device__ __forceinline__ void tdm_load_2d(const __bf16* src, unsigned lds_off,
                                            unsigned rows, unsigned cols,
                                            unsigned row_stride_elems) {
  unsigned long long ga = (unsigned long long)(uintptr_t)src;
  u32x4 g0;
  g0[0] = 1u;                                            // count=1, user mode
  g0[1] = lds_off;                                       // LDS byte address
  g0[2] = (unsigned)(ga & 0xffffffffull);                // global_addr[31:0]
  g0[3] = (unsigned)((ga >> 32) & 0x1ffffffull)          // global_addr[56:32]
        | 0x80000000u;                                   // type=2 ("image")
  i32x8 g1;
  g1[0] = (int)(1u << 16);                               // data_size code 1 = 2 bytes
  g1[1] = (int)((cols & 0xffffu) << 16);                 // tensor_dim0[15:0]
  g1[2] = (int)((cols >> 16) | ((rows & 0xffffu) << 16));// dim0[31:16] | dim1[15:0]
  g1[3] = (int)((rows >> 16) | (cols << 16));            // dim1[31:16] | tile_dim0
  g1[4] = (int)(rows);                                   // tile_dim1 (tile_dim2=0)
  g1[5] = (int)row_stride_elems;                         // tensor_dim0_stride[31:0]
  g1[6] = 0;                                             // stride0[47:32] | stride1 lo
  g1[7] = 0;
  i32x4 gz4 = {0, 0, 0, 0};
  i32x8 gz8 = {0, 0, 0, 0, 0, 0, 0, 0};
  __builtin_amdgcn_tensor_load_to_lds(g0, g1, gz4, gz4, gz8, 0);
}

// ---------------- prologue: weight prep ----------------
// Wcat[n][k] (bf16, [NCAT x KCAT]) :
//   rows 0..4095  = W_gates^T  (gate n, input k)
//   rows 4096..   = [W_w2d ; 0.5*W_h2d]^T  (alpha folded in)
__global__ void k_prep_wcat(const float* __restrict__ Wg,
                            const float* __restrict__ Ww2d,
                            const float* __restrict__ Wh2d,
                            __bf16* __restrict__ Wcat) {
  size_t idx = (size_t)blockIdx.x * 256 + threadIdx.x;
  size_t total = (size_t)NCAT * KCAT;
  if (idx >= total) return;
  int n = (int)(idx / KCAT);
  int k = (int)(idx % KCAT);
  float v;
  if (n < 4 * H_DIM) {
    v = Wg[(size_t)k * (4 * H_DIM) + n];
  } else {
    int j = n - 4 * H_DIM;
    v = (k < IN_DIM) ? Ww2d[(size_t)k * DA_DIM + j]
                     : 0.5f * Wh2d[(size_t)(k - IN_DIM) * DA_DIM + j];
  }
  Wcat[idx] = (__bf16)v;
}

// generic f32 [K x N] -> bf16 [N x K] transpose
__global__ void k_transpose_bf16(const float* __restrict__ src,
                                 __bf16* __restrict__ dst, int K, int N) {
  size_t idx = (size_t)blockIdx.x * 256 + threadIdx.x;
  size_t total = (size_t)K * N;
  if (idx >= total) return;
  int n = (int)(idx / K);
  int k = (int)(idx % K);
  dst[idx] = (__bf16)src[(size_t)k * N + n];
}

__global__ void k_cvt_bf16(const float* __restrict__ src,
                           __bf16* __restrict__ dst, size_t total) {
  size_t idx = (size_t)blockIdx.x * 256 + threadIdx.x;
  if (idx < total) dst[idx] = (__bf16)src[idx];
}

__global__ void k_init_state(const float* __restrict__ h0,
                             const float* __restrict__ C0,
                             const float* __restrict__ DA0,
                             float* __restrict__ Hst, __bf16* __restrict__ hbf,
                             float* __restrict__ Cst, float* __restrict__ DAst,
                             float* __restrict__ lossbuf) {
  size_t idx = (size_t)blockIdx.x * 256 + threadIdx.x;
  if (idx < 65536) {
    float v = h0[idx]; Hst[idx] = v; hbf[idx] = (__bf16)v;
  } else if (idx < 131072) {
    Cst[idx - 65536] = C0[idx - 65536];
  } else if (idx < 139264) {
    DAst[idx - 131072] = DA0[idx - 131072];
  } else if (idx == 139264) {
    lossbuf[0] = 0.0f;
  }
}

// ---------------- per-step kernel 1: gate+r GEMM ----------------
// ACT[b][n] = act( [x_t,h] @ Wcat^T ), n<4096 gates (sig/sig/sig/tanh), n>=4096 r=sigmoid
// grid: 66 blocks x 128 thr (4 waves); wave -> one N-tile, all 4 M-tiles.
// A-operand staged through LDS by the Tensor Data Mover, double-buffered.
__global__ void k_gates(const __bf16* __restrict__ xbf,   // [T,B,IN]
                        const __bf16* __restrict__ hbf,   // [B,H]
                        const __bf16* __restrict__ Wcat,  // [NCAT,KCAT]
                        float* __restrict__ ACT,          // [B,NCAT]
                        int t) {
  __shared__ __bf16 sA[2][BATCH * 128];                   // 2 x 16 KB K-chunks
  const int tid   = threadIdx.x;
  const int lane  = tid & 31;
  const int wave  = tid >> 5;
  const int ntile = blockIdx.x * 4 + wave;                // 0..263
  const int lr    = lane & 15;
  const int halfK = (lane >> 4) * 8;                      // lanes>=16 hold K+8

  v8f acc[4];
  for (int i = 0; i < 4; ++i) acc[i] = vzero8();

  const __bf16* xrow = xbf + (size_t)t * BATCH * IN_DIM;
  const __bf16* brow = Wcat + (size_t)(ntile * 16 + lr) * KCAT;

  // TDM issue for K-chunk kc into buffer buf (one wave drives the DMA).
  auto tdm_chunk = [&](int kc, int buf) {
    const int kbase = kc * 128;
    const __bf16* src = (kbase < IN_DIM) ? (xrow + kbase)
                                         : (hbf + (kbase - IN_DIM));
    unsigned stride = (kbase < IN_DIM) ? IN_DIM : H_DIM;
    unsigned lds_off = (unsigned)(uintptr_t)(&sA[buf][0]);
    tdm_load_2d(src, lds_off, BATCH, 128, stride);
  };

  if (wave == 0) tdm_chunk(0, 0);
  __builtin_amdgcn_s_wait_tensorcnt(0);
  __syncthreads();

  for (int kc = 0; kc < KCAT / 128; ++kc) {
    const int cur = kc & 1;
    if (wave == 0 && (kc + 1) < KCAT / 128) tdm_chunk(kc + 1, cur ^ 1);
    if ((kc + 1) < KCAT / 128)
      __builtin_prefetch(brow + (kc + 1) * 128, 0, 3);    // next B K-chunk, near scope
    const int kbase = kc * 128;
    for (int k32 = 0; k32 < 4; ++k32) {
      v16bf bfrag = load_frag(brow + kbase + k32 * 32 + halfK);
      v16bf afr[4];
      for (int mt = 0; mt < 4; ++mt)        // batch the 8 ds_load_b128s
        afr[mt] = load_frag(&sA[cur][(mt * 16 + lr) * 128 + k32 * 32 + halfK]);
      for (int mt = 0; mt < 4; ++mt)
        acc[mt] = __builtin_amdgcn_wmma_f32_16x16x32_bf16(
            false, afr[mt], false, bfrag, (short)0, acc[mt], false, false);
    }
    __builtin_amdgcn_s_wait_tensorcnt(0);                 // next chunk landed
    __syncthreads();
  }

  const int ncol = ntile * 16 + lr;
  const bool is_g = (ncol >= 3 * H_DIM) && (ncol < 4 * H_DIM);
  const int mo = (lane >> 4) * 8;
  for (int mt = 0; mt < 4; ++mt) {
    for (int vg = 0; vg < 8; ++vg) {
      int m = mt * 16 + vg + mo;
      float v = acc[mt][vg];
      ACT[(size_t)m * NCAT + ncol] = is_g ? tanhf(v) : sigmoidf_(v);
    }
  }
}

// ---------------- per-step kernel 2: DA update + norm + loss ----------------
__global__ void k_da(const float* __restrict__ ACT,
                     float* __restrict__ DAst, __bf16* __restrict__ DAbf,
                     float* __restrict__ lossbuf) {
  __shared__ float red[256];
  const int tid = threadIdx.x;
  float sq = 0.f;
  for (int i = 0; i < 32; ++i) {
    int e = tid + i * 256;                  // 0..8191
    int m = e >> 7, j = e & 127;
    float r  = ACT[(size_t)m * NCAT + 4 * H_DIM + j];  // already sigmoid'ed
    float da = DAst[e];
    float dn = r * da;
    float d  = dn - da;
    sq += d * d;
    DAst[e] = dn;
    DAbf[e] = (__bf16)dn;
  }
  red[tid] = sq;
  __syncthreads();
  for (int s = 128; s > 0; s >>= 1) {
    if (tid < s) red[tid] += red[tid + s];
    __syncthreads();
  }
  if (tid == 0) {
    float st = sqrtf(red[0]);
    lossbuf[0] += ETA_C * __powf(KEXI_C, st);
  }
}

// ---------------- per-step kernel 3: cell GEMM + elementwise ----------------
// C_new = f*C + i*g + tanh(DA_new @ W_d2c);  h_new = o*tanh(C_new)
// grid: 16 blocks x 128 thr; wave -> 1 N-tile of 16 cols, all 4 M-tiles.
__global__ void k_cell(const __bf16* __restrict__ DAbf,   // [B,DA]
                       const __bf16* __restrict__ Wd2cT,  // [H,DA]
                       const float* __restrict__ ACT,
                       float* __restrict__ Cst,           // [B,H]
                       float* __restrict__ Hst,           // [B,H]
                       __bf16* __restrict__ hbf,          // [B,H]
                       __bf16* __restrict__ hsbf,         // [T,B,H]
                       int t) {
  const int tid   = threadIdx.x;
  const int lane  = tid & 31;
  const int wave  = tid >> 5;
  const int ntile = blockIdx.x * 4 + wave;                // 0..63
  const int lr    = lane & 15;
  const int halfK = (lane >> 4) * 8;

  v8f acc[4];
  for (int i = 0; i < 4; ++i) acc[i] = vzero8();

  for (int k32 = 0; k32 < DA_DIM / 32; ++k32) {
    v16bf bfrag = load_frag(Wd2cT + (size_t)(ntile * 16 + lr) * DA_DIM
                                  + k32 * 32 + halfK);
    v16bf afr[4];
    for (int mt = 0; mt < 4; ++mt)
      afr[mt] = load_frag(DAbf + (size_t)(mt * 16 + lr) * DA_DIM
                               + k32 * 32 + halfK);
    for (int mt = 0; mt < 4; ++mt)
      acc[mt] = __builtin_amdgcn_wmma_f32_16x16x32_bf16(
          false, afr[mt], false, bfrag, (short)0, acc[mt], false, false);
  }

  const int n  = ntile * 16 + lr;
  const int mo = (lane >> 4) * 8;
  for (int mt = 0; mt < 4; ++mt) {
    for (int vg = 0; vg < 8; ++vg) {
      int m = mt * 16 + vg + mo;
      size_t ab = (size_t)m * NCAT;
      float iG = ACT[ab + n];
      float fG = ACT[ab + H_DIM + n];
      float oG = ACT[ab + 2 * H_DIM + n];
      float gG = ACT[ab + 3 * H_DIM + n];
      size_t hx = (size_t)m * H_DIM + n;
      float c  = fG * Cst[hx] + iG * gG + tanhf(acc[mt][vg]);
      Cst[hx]  = c;
      float h  = oG * tanhf(c);
      Hst[hx]  = h;
      __bf16 hb = (__bf16)h;
      hbf[hx] = hb;
      hsbf[((size_t)t * BATCH + m) * H_DIM + n] = hb;
    }
  }
}

// ---------------- final projection: hs @ W_h2o, (t,b) -> (b,t) transpose ----------------
// grid: 8192 blocks x 128 thr. block = (mblk of 64 rows, nblk of 64 cols).
__global__ void k_proj(const __bf16* __restrict__ hsbf,   // [T*B, H]
                       const __bf16* __restrict__ Wh2oT,  // [V, H]
                       float* __restrict__ out) {
  const int tid   = threadIdx.x;
  const int lane  = tid & 31;
  const int wave  = tid >> 5;
  const int mblk  = blockIdx.x & 255;                     // 256 blocks of 64 rows
  const int nblk  = blockIdx.x >> 8;                      // 32 blocks of 64 cols
  const int ntile = nblk * 4 + wave;                      // 0..127
  const int lr    = lane & 15;
  const int halfK = (lane >> 4) * 8;

  v8f acc[4];
  for (int i = 0; i < 4; ++i) acc[i] = vzero8();

  for (int k32 = 0; k32 < H_DIM / 32; ++k32) {
    v16bf bfrag = load_frag(Wh2oT + (size_t)(ntile * 16 + lr) * H_DIM
                                  + k32 * 32 + halfK);
    v16bf afr[4];
    for (int mt = 0; mt < 4; ++mt)
      afr[mt] = load_frag(hsbf + (size_t)(mblk * 64 + mt * 16 + lr) * H_DIM
                               + k32 * 32 + halfK);
    for (int mt = 0; mt < 4; ++mt)
      acc[mt] = __builtin_amdgcn_wmma_f32_16x16x32_bf16(
          false, afr[mt], false, bfrag, (short)0, acc[mt], false, false);
  }

  const int n  = ntile * 16 + lr;
  const int mo = (lane >> 4) * 8;
  for (int mt = 0; mt < 4; ++mt) {
    for (int vg = 0; vg < 8; ++vg) {
      int m  = mblk * 64 + mt * 16 + vg + mo;             // row = t*64 + b
      int tt = m >> 6;
      int b  = m & 63;
      out[((size_t)b * T_STEPS + tt) * V_DIM + n] = acc[mt][vg];
    }
  }
}

// ---------------- epilogue: final states + loss ----------------
__global__ void k_tail(const float* __restrict__ Hst, const float* __restrict__ Cst,
                       const float* __restrict__ DAst, const float* __restrict__ lossbuf,
                       float* __restrict__ out) {
  size_t idx = (size_t)blockIdx.x * 256 + threadIdx.x;
  float* tail = out + (size_t)BATCH * T_STEPS * V_DIM;
  if (idx < 65536)        tail[idx] = Hst[idx];
  else if (idx < 131072)  tail[idx] = Cst[idx - 65536];
  else if (idx < 139264)  tail[idx] = DAst[idx - 131072];
  else if (idx == 139264) tail[idx] = lossbuf[0];
}

// ---------------- host launch ----------------
extern "C" void kernel_launch(void* const* d_in, const int* in_sizes, int n_in,
                              void* d_out, int out_size, void* d_ws, size_t ws_size,
                              hipStream_t stream) {
  (void)in_sizes; (void)n_in; (void)out_size; (void)ws_size;
  const float* input   = (const float*)d_in[0];
  const float* h0      = (const float*)d_in[1];
  const float* C0      = (const float*)d_in[2];
  const float* DA0     = (const float*)d_in[3];
  const float* Wg      = (const float*)d_in[4];
  const float* Ww2d    = (const float*)d_in[5];
  const float* Wh2d    = (const float*)d_in[6];
  const float* Wd2c_in = (const float*)d_in[7];
  const float* Wh2o_in = (const float*)d_in[8];
  float* out = (float*)d_out;

  // ---- ws bump allocator (256B aligned) ----
  char* wsp = (char*)d_ws;
  auto alloc = [&](size_t bytes) -> void* {
    void* p = (void*)wsp;
    wsp += (bytes + 255) & ~(size_t)255;
    return p;
  };
  __bf16* Wcat   = (__bf16*)alloc((size_t)NCAT * KCAT * 2);            // 12.98 MB
  __bf16* Wd2cT  = (__bf16*)alloc((size_t)H_DIM * DA_DIM * 2);         // 0.26 MB
  __bf16* Wh2oT  = (__bf16*)alloc((size_t)V_DIM * H_DIM * 2);          // 4.19 MB
  __bf16* xbf    = (__bf16*)alloc((size_t)T_STEPS * BATCH * IN_DIM * 2);   // 16.8 MB
  __bf16* hsbf   = (__bf16*)alloc((size_t)T_STEPS * BATCH * H_DIM * 2);    // 33.6 MB
  float*  ACT    = (float*) alloc((size_t)BATCH * NCAT * 4);           // 1.08 MB
  float*  Hst    = (float*) alloc((size_t)BATCH * H_DIM * 4);
  float*  Cst    = (float*) alloc((size_t)BATCH * H_DIM * 4);
  float*  DAst   = (float*) alloc((size_t)BATCH * DA_DIM * 4);
  __bf16* hbf    = (__bf16*)alloc((size_t)BATCH * H_DIM * 2);
  __bf16* DAbf   = (__bf16*)alloc((size_t)BATCH * DA_DIM * 2);
  float*  lossbuf= (float*) alloc(256);

  // ---- prologue ----
  {
    size_t tot = (size_t)NCAT * KCAT;
    k_prep_wcat<<<(unsigned)((tot + 255) / 256), 256, 0, stream>>>(Wg, Ww2d, Wh2d, Wcat);
  }
  k_transpose_bf16<<<(H_DIM * DA_DIM + 255) / 256, 256, 0, stream>>>(Wd2c_in, Wd2cT, DA_DIM, H_DIM);
  k_transpose_bf16<<<(V_DIM * H_DIM + 255) / 256, 256, 0, stream>>>(Wh2o_in, Wh2oT, H_DIM, V_DIM);
  {
    size_t tot = (size_t)T_STEPS * BATCH * IN_DIM;
    k_cvt_bf16<<<(unsigned)((tot + 255) / 256), 256, 0, stream>>>(input, xbf, tot);
  }
  k_init_state<<<(139265 + 255) / 256, 256, 0, stream>>>(h0, C0, DA0, Hst, hbf, Cst, DAst, lossbuf);

  // ---- recurrence ----
  for (int t = 0; t < T_STEPS; ++t) {
    k_gates<<<66, 128, 0, stream>>>(xbf, hbf, Wcat, ACT, t);
    k_da<<<1, 256, 0, stream>>>(ACT, DAst, DAbf, lossbuf);
    k_cell<<<16, 128, 0, stream>>>(DAbf, Wd2cT, ACT, Cst, Hst, hbf, hsbf, t);
  }

  // ---- final projection + tail ----
  k_proj<<<8192, 128, 0, stream>>>(hsbf, Wh2oT, out);
  k_tail<<<(139265 + 255) / 256, 256, 0, stream>>>(Hst, Cst, DAst, lossbuf, out);
}
